// MultiheadAttn_35210141892613
// MI455X (gfx1250) — matrix operand order repaired and checked by reading, not compiled, not run on hardware
//
#include <hip/hip_runtime.h>
#include <hip/hip_bf16.h>

// ---------------------------------------------------------------------------
// MultiheadAttn (graph attention, aggtype='mean') for MI455X / gfx1250.
// N=30000 nodes, K=32 neighbors, D=128 features, H=8 heads.
//
// Pipeline:
//   k0: one-time f32 -> bf16 weight/activation pre-conversion
//   k1: p = relu(prev @ fc1^T)            [WMMA bf16; p stored f32 + bf16]
//       xh = p @ W1_all^T (1024 cols)     [WMMA bf16; stored bf16]
//   k2: per-node attention on the matrix pipe:
//         scores = xh(8x128) @ neigh^T    [8 WMMA/node]
//         softmax (wave32 shuffles)
//         agg    = attn(8x32) @ neigh     [8 WMMA/node, strided B gather]
//       neigh staged with global_load_async_to_lds_b128 (ASYNCcnt)
//   k3: pre_h = agg_h @ A1_h^T + p @ A2_h^T   [8 WMMA per col tile, fused]
//       tanh, mean over heads (ds_add_f32), elu, concat(p,.), relu
//
// Traffic: neigh 491MB + xh/agg/p round-trips ~ 860MB total  (~37us @23.3TB/s)
// ---------------------------------------------------------------------------

#define NN     30000
#define DD     128
#define KNB    32
#define HH     8
#define ELU_A  0.2f

typedef __bf16 bf16;
typedef __attribute__((ext_vector_type(16))) __bf16 v16bf;
typedef __attribute__((ext_vector_type(8)))  float  v8f;

// ---------------------------------------------------------------------------
// WMMA helpers.  V_WMMA_F32_16X16X32_BF16 operand layouts (ISA 7.12.2):
//   A (16x32, MxK): lane&15 = row M; lane<16 -> K 0..15 (order used here),
//                   lane>=16 -> K 16..31. 16 bf16 per lane.
//   B (32x16, KxN): lane&15 = col N; same K order as A (this code's choice; ISA table differs).
//   C/D (16x16):    elem v -> M = v + 8*(lane>>4), N = lane&15.
// ---------------------------------------------------------------------------

__device__ __forceinline__ v8f wmma_bf16(v16bf a, v16bf b, v8f c) {
  return __builtin_amdgcn_wmma_f32_16x16x32_bf16(
      /*neg_a=*/false, a, /*neg_b=*/false, b,
      /*c_mod=*/(short)0, c, /*reuse_a=*/false, /*reuse_b=*/false);
}

// A-fragment from two contiguous 8-bf16 (16B) runs
__device__ __forceinline__ v16bf fragA_bf16(const bf16* r0, const bf16* r1) {
  v16bf a;
#pragma unroll
  for (int j = 0; j < 8; ++j) a[j] = r0[j];
#pragma unroll
  for (int j = 0; j < 8; ++j) a[8 + j] = r1[j];
  return a;
}

// B-fragment: one contiguous 16-bf16 (32B) run
__device__ __forceinline__ v16bf fragB_bf16(const bf16* r) {
  v16bf b;
#pragma unroll
  for (int j = 0; j < 16; ++j) b[j] = r[j];
  return b;
}

// B-fragment: one contiguous 16-f32 (64B) run, convert to bf16
__device__ __forceinline__ v16bf fragB_f32cvt(const float* r) {
  v16bf b;
#pragma unroll
  for (int j = 0; j < 16; ++j) b[j] = (bf16)r[j];
  return b;
}

// CDNA5 async global -> LDS copy (ASYNCcnt tracked, no VGPR round-trip).
// VGLOBAL form: global_load_async_to_lds_b128 vdst(lds byte addr), vaddr, off
__device__ __forceinline__ void async_ld_b128(void* lds_ptr, const void* gptr) {
  unsigned int       l = (unsigned int)(unsigned long long)(uintptr_t)lds_ptr;
  unsigned long long g = (unsigned long long)(uintptr_t)gptr;
  asm volatile("global_load_async_to_lds_b128 %0, %1, off"
               :: "v"(l), "v"(g) : "memory");
}
__device__ __forceinline__ void async_wait0() {
  asm volatile("s_wait_asynccnt 0x0" ::: "memory");
}

// ---------------------------------------------------------------------------
// Kernel 0: one-time f32 -> bf16 conversions (weights stay hot in 192MB L2).
//   fc1b  [128][128]
//   wcatb [2048][128] : rows 0..1023 = w1 flat (xh GEMM, k1)
//                       rows 1024..2047 = a_w[:,:,128:] (A2, p-part GEMM, k3)
//   a1b   [1024][128] : a_w[:,:,:128]  (A1, agg-part GEMM, k3)
//   prevb [N][128]
// ---------------------------------------------------------------------------
__global__ void k0_cvt(const float* __restrict__ fc1, const float* __restrict__ w1,
                       const float* __restrict__ aw,  const float* __restrict__ prev,
                       bf16* __restrict__ fc1b, bf16* __restrict__ wcatb,
                       bf16* __restrict__ a1b,  bf16* __restrict__ prevb) {
  const size_t tid = (size_t)blockIdx.x * blockDim.x + threadIdx.x;
  const size_t stride = (size_t)gridDim.x * blockDim.x;
  for (size_t j = tid; j < 128 * 128; j += stride) fc1b[j] = (bf16)fc1[j];
  for (size_t j = tid; j < 1024 * 128; j += stride) wcatb[j] = (bf16)w1[j];
  for (size_t j = tid; j < 1024 * 128; j += stride)
    wcatb[1024 * 128 + j] = (bf16)aw[(j >> 7) * 256 + 128 + (j & 127)];
  for (size_t j = tid; j < 1024 * 128; j += stride)
    a1b[j] = (bf16)aw[(j >> 7) * 256 + (j & 127)];
  for (size_t j = tid; j < (size_t)NN * 128; j += stride) prevb[j] = (bf16)prev[j];
}

// ---------------------------------------------------------------------------
// Kernel 1: fused linear stage.  grid = N/16 blocks, 256 threads (8 waves).
// ---------------------------------------------------------------------------
#define PST 136

__global__ void k1_linear(const bf16* __restrict__ prevb,
                          const bf16* __restrict__ fc1b,
                          const bf16* __restrict__ wcatb,
                          float* __restrict__ p_out,
                          bf16*  __restrict__ pb_out,
                          bf16*  __restrict__ xh_out) {
  extern __shared__ float smem[];
  bf16* p_lds = (bf16*)smem;

  const int wave = threadIdx.x >> 5;
  const int lane = threadIdx.x & 31;
  const int m0   = blockIdx.x * 16;

  const int row   = lane & 15;
  const int kb    = (lane >> 4) * 8;
  const int khalf = (lane >> 4) * 16;
  const int colL  = lane & 15;
  const int mrow  = (lane >> 4) * 8;

  // ---- stage 1: p tile, wave w computes cols [16w, 16w+16) -----------------
  {
    const bf16* Arow = prevb + (size_t)(m0 + row) * DD;
    const bf16* Brow = fc1b + (size_t)(wave * 16 + colL) * DD;
    v8f acc = {};
#pragma unroll
    for (int k0 = 0; k0 < DD; k0 += 32) {
      v16bf a = fragA_bf16(Arow + k0 + khalf, Arow + k0 + khalf + 8);
      v16bf b = fragB_bf16(Brow + k0 + khalf);
      acc = wmma_bf16(a, b, acc);
    }
#pragma unroll
    for (int v = 0; v < 8; ++v) {
      int m = v + mrow;
      float val = fmaxf(acc[v], 0.0f);                 // relu
      p_out[(size_t)(m0 + m) * DD + wave * 16 + colL]  = val;
      pb_out[(size_t)(m0 + m) * DD + wave * 16 + colL] = (bf16)val;
      p_lds[m * PST + wave * 16 + colL] = (bf16)val;
    }
  }
  __syncthreads();

  // ---- stage 2: xh = p @ W1_all^T, 1024 cols (8 col-tiles per wave) --------
  v16bf afr[4];
#pragma unroll
  for (int i = 0; i < 4; ++i) {
    const bf16* r = p_lds + row * PST + i * 32;
    afr[i] = fragA_bf16(r + khalf, r + khalf + 8);
  }

#pragma unroll 1
  for (int t = 0; t < 8; ++t) {
    const int ct   = wave * 8 + t;          // col tile 0..63
    const int c    = ct * 16 + colL;
    const bf16* Brow = wcatb + (size_t)c * DD;
    v8f acc = {};
#pragma unroll
    for (int i = 0; i < 4; ++i) {
      v16bf b = fragB_bf16(Brow + i * 32 + khalf);
      acc = wmma_bf16(afr[i], b, acc);
    }
#pragma unroll
    for (int v = 0; v < 8; ++v) {
      int m = v + mrow;
      xh_out[(size_t)(m0 + m) * 1024 + c] = (bf16)acc[v];
    }
  }
}

// ---------------------------------------------------------------------------
// Kernel 2: per-node attention on the matrix pipe.  grid = N/8 blocks,
// 256 threads (8 waves), one node per wave.  ~165 KB dynamic LDS / WG.
// Per-wave LDS regions (padded strides for conflict-free access):
//   nb  [32][132] f32   async-staged neigh tile (row-major)
//   xl  [8][136]  bf16  xh rows (A operand for scores)
//   sc  [8][33]   f32   raw score tiles (C-layout -> lane-per-k handoff)
//   at  [8][34]   bf16  softmaxed attention (A operand for agg)
// ---------------------------------------------------------------------------
#define NBST2 132
#define XLST  136
#define SCST  33
#define ATST  34

__global__ void k2_attn(const float* __restrict__ neigh,
                        const bf16* __restrict__ xh,
                        bf16* __restrict__ agg_out) {
  extern __shared__ float smem[];
  const int wave = threadIdx.x >> 5;
  const int lane = threadIdx.x & 31;
  const int n    = blockIdx.x * 8 + wave;

  const size_t NB_SZ = KNB * NBST2 * 4;   // 16896
  const size_t XL_SZ = HH * XLST * 2;     // 2176
  const size_t SC_SZ = HH * SCST * 4;     // 1056
  const size_t AT_SZ = HH * ATST * 2;     // 544
  const size_t WV_SZ = NB_SZ + XL_SZ + SC_SZ + AT_SZ;   // 20672 (16B mult)

  char* wbase = (char*)smem + (size_t)wave * WV_SZ;
  float* nb = (float*)wbase;
  bf16*  xl = (bf16*)(wbase + NB_SZ);
  float* sc = (float*)(wbase + NB_SZ + XL_SZ);
  bf16*  at = (bf16*)(wbase + NB_SZ + XL_SZ + SC_SZ);

  // ---- async stage: neigh[n] 32x128 f32 (32 b128/lane), xh[n] 1024 bf16 ----
  const float* src = neigh + (size_t)n * KNB * DD;
  __builtin_prefetch(src + 8 * KNB * DD, 0, 0);
#pragma unroll
  for (int i = 0; i < 32; ++i) {
    int elem = (i * 32 + lane) * 4;
    async_ld_b128(nb + (elem >> 7) * NBST2 + (elem & 127), src + elem);
  }
  const bf16* xsrc = xh + (size_t)n * 1024;
#pragma unroll
  for (int i = 0; i < 4; ++i) {
    int elem = (i * 32 + lane) * 8;
    async_ld_b128(xl + (elem >> 7) * XLST + (elem & 127), xsrc + elem);
  }
  async_wait0();
  __syncthreads();

  const int rowm  = lane & 15;
  const int kb    = (lane >> 4) * 8;
  const int khalf = (lane >> 4) * 16;
  const int colL  = lane & 15;
  const v16bf zfr = {};

  // ---- scores = xh(8x128) @ neigh^T(128x32): rows 8..15 zero ---------------
  v16bf afr[4];
#pragma unroll
  for (int i = 0; i < 4; ++i) {
    if (rowm < HH) {
      const bf16* r = xl + rowm * XLST + i * 32;
      afr[i] = fragA_bf16(r + khalf, r + khalf + 8);
    } else {
      afr[i] = zfr;
    }
  }
#pragma unroll
  for (int t = 0; t < 2; ++t) {
    // B[kd][col] = neigh[col][kd]: contiguous 16-f32 run per lane
    const float* Brow = nb + (t * 16 + colL) * NBST2;
    v8f acc = {};
#pragma unroll
    for (int i = 0; i < 4; ++i)
      acc = wmma_bf16(afr[i], fragB_f32cvt(Brow + i * 32 + khalf), acc);
    if (lane < 16) {          // rows 0..7 live in elems 0..7 of lanes 0..15
#pragma unroll
      for (int v = 0; v < HH; ++v) sc[v * SCST + t * 16 + colL] = acc[v];
    }
  }
  __syncthreads();

  // ---- softmax over k (lane = k), write bf16 attn in A-row layout ----------
#pragma unroll
  for (int h = 0; h < HH; ++h) {
    float sv = sc[h * SCST + lane];
    float m = sv;
#pragma unroll
    for (int off = 16; off > 0; off >>= 1) m = fmaxf(m, __shfl_xor(m, off, 32));
    float e = __expf(sv - m);
    float sum = e;
#pragma unroll
    for (int off = 16; off > 0; off >>= 1) sum += __shfl_xor(sum, off, 32);
    at[h * ATST + lane] = (bf16)(e / sum);
  }
  __syncthreads();

  // ---- agg = attn(8x32) @ neigh(32x128): K=32 -> 1 WMMA per 16-col tile ----
  // B[k][d] gathered directly from row-major f32 nb (strided ds_load + cvt).
  v16bf aat;
  if (rowm < HH) {
    const bf16* r = at + rowm * ATST;
    aat = fragA_bf16(r + khalf, r + khalf + 8);
  } else {
    aat = zfr;
  }
  bf16* dst = agg_out + (size_t)n * HH * DD;
#pragma unroll 1
  for (int ct = 0; ct < 8; ++ct) {
    const int d = ct * 16 + colL;
    v16bf b;
#pragma unroll
    for (int j = 0; j < 16; ++j)
      b[j] = (bf16)nb[(khalf + j) * NBST2 + d];
    v8f zc = {};
    v8f acc = wmma_bf16(aat, b, zc);
    if (lane < 16) {
#pragma unroll
      for (int v = 0; v < HH; ++v)
        dst[v * DD + ct * 16 + colL] = (bf16)acc[v];
    }
  }
}

// ---------------------------------------------------------------------------
// Kernel 3: output head.  grid = N/16 blocks, 256 threads; wave w owns head w.
//   pre_h = agg_h @ A1_h^T + p @ A2_h^T  (8 WMMA per col tile, fused in C)
//   tanh ; ds_add_f32 reduce over heads ; mean, elu, concat(p,.), relu.
// ---------------------------------------------------------------------------
__global__ void k3_out(const bf16* __restrict__ agg,
                       const bf16* __restrict__ pb,
                       const float* __restrict__ p,
                       const bf16* __restrict__ a1b,
                       const bf16* __restrict__ wcatb,
                       float* __restrict__ out) {
  __shared__ float accum[16][132];
  const int wave = threadIdx.x >> 5;   // head
  const int lane = threadIdx.x & 31;
  const int m0   = blockIdx.x * 16;
  const int h    = wave;

  for (int i = threadIdx.x; i < 16 * 132; i += 256) ((float*)accum)[i] = 0.0f;
  __syncthreads();

  const int row   = lane & 15;
  const int kb    = (lane >> 4) * 8;
  const int khalf = (lane >> 4) * 16;
  const int colL  = lane & 15;
  const int mrow  = (lane >> 4) * 8;

  // A fragments: agg[m0+row, h, :] and p_bf[m0+row, :]  — tile-invariant
  const bf16* Ab1 = agg + ((size_t)(m0 + row) * HH + h) * DD;
  const bf16* Ab2 = pb + (size_t)(m0 + row) * DD;
  v16bf af1[4], af2[4];
#pragma unroll
  for (int i = 0; i < 4; ++i) {
    af1[i] = fragA_bf16(Ab1 + i * 32 + khalf, Ab1 + i * 32 + khalf + 8);
    af2[i] = fragA_bf16(Ab2 + i * 32 + khalf, Ab2 + i * 32 + khalf + 8);
  }

#pragma unroll 1
  for (int ct = 0; ct < 8; ++ct) {
    const int col0 = ct * 16;
    const bf16* Brow1 = a1b + (size_t)(h * DD + col0 + colL) * DD;            // A1_h
    const bf16* Brow2 = wcatb + (size_t)(1024 + h * DD + col0 + colL) * DD;   // A2_h
    v8f acc = {};
#pragma unroll
    for (int i = 0; i < 4; ++i)
      acc = wmma_bf16(af1[i], fragB_bf16(Brow1 + i * 32 + khalf), acc);
#pragma unroll
    for (int i = 0; i < 4; ++i)
      acc = wmma_bf16(af2[i], fragB_bf16(Brow2 + i * 32 + khalf), acc);
#pragma unroll
    for (int v = 0; v < 8; ++v) {
      int m = v + mrow;
      int e = col0 + colL;
      float val = tanhf(acc[v]);
      atomicAdd(&accum[m][e], val);     // ds_add_f32 across the 8 heads
    }
  }
  __syncthreads();

  for (int i = threadIdx.x; i < 16 * DD; i += 256) {
    int m = i >> 7, e = i & 127;
    float mean = accum[m][e] * 0.125f;
    float elu  = (mean > 0.0f) ? mean : ELU_A * expm1f(mean);
    size_t ob  = (size_t)(m0 + m) * 256;
    out[ob + e]       = fmaxf(p[(size_t)(m0 + m) * DD + e], 0.0f);
    out[ob + 128 + e] = fmaxf(elu, 0.0f);
  }
}

// ---------------------------------------------------------------------------
// Launch.  Workspace: p f32 (15.4MB) | pb bf16 (7.7MB) | xh bf16 (61.4MB) |
//                     agg bf16 (61.4MB) | prevb bf16 (7.7MB) | weights (~0.8MB)
// ---------------------------------------------------------------------------
extern "C" void kernel_launch(void* const* d_in, const int* in_sizes, int n_in,
                              void* d_out, int out_size, void* d_ws, size_t ws_size,
                              hipStream_t stream) {
  const float* prev  = (const float*)d_in[0];
  const float* neigh = (const float*)d_in[1];
  const float* fc1   = (const float*)d_in[2];
  const float* w1    = (const float*)d_in[3];
  const float* aw    = (const float*)d_in[4];
  float* out = (float*)d_out;

  char* ws = (char*)d_ws;
  size_t off = 0;
  float* p_ws   = (float*)(ws + off); off += (size_t)NN * DD * sizeof(float);
  bf16*  pb_ws  = (bf16*)(ws + off);  off += (size_t)NN * DD * sizeof(bf16);
  bf16*  xh_ws  = (bf16*)(ws + off);  off += (size_t)NN * 1024 * sizeof(bf16);
  bf16*  agg_ws = (bf16*)(ws + off);  off += (size_t)NN * 1024 * sizeof(bf16);
  bf16*  prevb  = (bf16*)(ws + off);  off += (size_t)NN * DD * sizeof(bf16);
  bf16*  fc1b   = (bf16*)(ws + off);  off += (size_t)128 * 128 * sizeof(bf16);
  bf16*  wcatb  = (bf16*)(ws + off);  off += (size_t)2048 * 128 * sizeof(bf16);
  bf16*  a1b    = (bf16*)(ws + off);  off += (size_t)1024 * 128 * sizeof(bf16);

  const size_t smem1 = 16 * PST * sizeof(bf16);
  const size_t smem2 = (size_t)8 *
      (KNB * NBST2 * 4 + HH * XLST * 2 + HH * SCST * 4 + HH * ATST * 2);

  k0_cvt   <<<512,     256, 0,     stream>>>(fc1, w1, aw, prev, fc1b, wcatb, a1b, prevb);
  k1_linear<<<NN / 16, 256, smem1, stream>>>(prevb, fc1b, wcatb, p_ws, pb_ws, xh_ws);
  k2_attn  <<<NN / 8,  256, smem2, stream>>>(neigh, xh_ws, agg_ws);
  k3_out   <<<NN / 16, 256, 0,     stream>>>(agg_ws, pb_ws, p_ws, a1b, wcatb, out);
}
